// Transformer_58497454571543
// MI455X (gfx1250) — matrix operand order repaired and checked by reading, not compiled, not run on hardware
//
#include <hip/hip_runtime.h>
#include <hip/hip_bf16.h>

typedef __attribute__((ext_vector_type(16))) _Float16 v16h;
typedef __attribute__((ext_vector_type(8)))  _Float16 v8h;
typedef __attribute__((ext_vector_type(8)))  float    v8f;

#define D_MODEL 1024
#define NHEAD   16
#define DHEAD   64
#define SEQ     512
#define BATCH   2
#define MROWS   (BATCH * SEQ)   // 1024
#define VOCAB   32000
#define NLAYERS 4

enum { LAY_KN = 0, LAY_NK = 1, LAY_HDE = 2 };

// convert 8 consecutive f32 -> f16 and store as one 16-byte LDS write
__device__ __forceinline__ void cvt_store8(_Float16* __restrict__ dst,
                                           const float* __restrict__ src)
{
    const float4 a = *(const float4*)src;
    const float4 b = *(const float4*)(src + 4);
    v8h h;
    h[0] = (_Float16)a.x; h[1] = (_Float16)a.y;
    h[2] = (_Float16)a.z; h[3] = (_Float16)a.w;
    h[4] = (_Float16)b.x; h[5] = (_Float16)b.y;
    h[6] = (_Float16)b.z; h[7] = (_Float16)b.w;
    *(v8h*)dst = h;
}

// ---------------------------------------------------------------------------
// Generic WMMA GEMM:  Out[M,N] = act(A[M,K] @ B + bias)
//   A: f32 row-major, leading dim ldA (staged to f16 in LDS)
//   B layouts: LAY_KN  -> B[k*ldB + n]
//              LAY_NK  -> B[n*ldB + k]            (B given transposed)
//              LAY_HDE -> qkv weight [H, D, DH]:  head = n/64 block
//   Batched over blockIdx.z, z = b*Hn + h.
//   STREAM: non-temporal output stores (final 131MB vocab projection).
// Block: 128 threads = 4 waves; block tile 128(M) x 64(N);
// each wave owns 32(M) x 64(N) = 8 WMMA accumulators (B frags reused 2x).
// ---------------------------------------------------------------------------
template <int BLAY, bool RELU, bool HASBIAS, bool STREAM>
__global__ __launch_bounds__(128) void wmma_gemm(
    const float* __restrict__ A, const float* __restrict__ Bw,
    const float* __restrict__ bias, float* __restrict__ Out,
    int M, int N, int K, int ldA, int ldB, int ldOut, int Hn,
    long long sAb, long long sAh, long long sBb, long long sBh,
    long long sOb, long long sOh)
{
    // rows padded to 48 halves (96 B) so 16B (v8h) and 32B (v16h) LDS
    // accesses stay naturally aligned.
    __shared__ __align__(32) _Float16 As[128][48];
    __shared__ __align__(32) _Float16 Bt[64][48];

    const int tid  = threadIdx.x;
    const int wv   = tid >> 5;
    const int lane = tid & 31;
    const int g    = lane >> 4;
    const int nl   = lane & 15;

    const int n0 = blockIdx.x * 64;
    const int m0 = blockIdx.y * 128;
    const int z  = blockIdx.z;
    const int bz = z / Hn;
    const int hz = z % Hn;

    const size_t aOff = (size_t)bz * sAb + (size_t)hz * sAh;
    size_t       bOff = (size_t)bz * sBb + (size_t)hz * sBh;
    const size_t oOff = (size_t)bz * sOb + (size_t)hz * sOh;

    int nBase = n0;
    if (BLAY == LAY_HDE) {
        // weight[h][d][e]: a 64-wide N tile lies entirely inside head n0>>6
        bOff += (size_t)(n0 >> 6) * (size_t)K * 64;
        nBase = 0;
    }

    v8f acc[2][4] = {};

    for (int k0 = 0; k0 < K; k0 += 32) {
        // ---- prefetch next k-step tiles (global_prefetch_b8) ----
        const int kn = k0 + 32;
        if (kn < K) {
            __builtin_prefetch(A + aOff + (size_t)(m0 + tid) * ldA + kn, 0, 1);
            if (BLAY == LAY_NK)
                __builtin_prefetch(Bw + bOff + (size_t)(n0 + (tid >> 1)) * ldB + kn, 0, 1);
            else
                __builtin_prefetch(Bw + bOff + (size_t)(kn + (tid >> 2)) * ldB + nBase, 0, 1);
        }

        // ---- stage A tile 128x32 (f32 -> f16), vectorized ----
        {
            const int cb = (tid & 1) << 4;
            #pragma unroll
            for (int rr = 0; rr < 2; ++rr) {
                const int r = (tid >> 1) + (rr << 6);
                const float* ap = A + aOff + (size_t)(m0 + r) * ldA + (k0 + cb);
                cvt_store8(&As[r][cb], ap);
                cvt_store8(&As[r][cb + 8], ap + 8);
            }
        }
        // ---- stage B tile, transposed to Bt[n][k] ----
        if (BLAY == LAY_NK) {
            const int n  = tid >> 1;
            const int kb = (tid & 1) << 4;
            const float* bp = Bw + bOff + (size_t)(n0 + n) * ldB + (k0 + kb);
            cvt_store8(&Bt[n][kb], bp);
            cvt_store8(&Bt[n][kb + 8], bp + 8);
        } else {
            // transpose via (k,k+1) half-pair packing -> aligned b32 LDS stores
            const int kp = tid >> 3;            // 0..15  (k = 2*kp)
            const int nb = (tid & 7) << 3;      // 0,8,...,56
            const float* b0 = Bw + bOff + (size_t)(k0 + 2 * kp) * ldB + (nBase + nb);
            const float* b1 = b0 + ldB;
            const float4 x0 = *(const float4*)b0;
            const float4 x1 = *(const float4*)(b0 + 4);
            const float4 y0 = *(const float4*)b1;
            const float4 y1 = *(const float4*)(b1 + 4);
            const float r0[8] = {x0.x, x0.y, x0.z, x0.w, x1.x, x1.y, x1.z, x1.w};
            const float r1[8] = {y0.x, y0.y, y0.z, y0.w, y1.x, y1.y, y1.z, y1.w};
            #pragma unroll
            for (int n = 0; n < 8; ++n) {
                union { _Float16 h[2]; unsigned u; } p;
                p.h[0] = (_Float16)r0[n];
                p.h[1] = (_Float16)r1[n];
                *(unsigned*)&Bt[nb + n][2 * kp] = p.u;
            }
        }
        __syncthreads();

        // ---- two A fragments per wave (rows wv*32 .. wv*32+31) ----
        v16h af[2];
        #pragma unroll
        for (int s = 0; s < 2; ++s) {
            const _Float16* ap = &As[(wv << 5) + (s << 4) + nl][0];
            v8h lo = *(const v8h*)(ap + (g << 4));
            v8h hi = *(const v8h*)(ap + 8 + (g << 4));
            #pragma unroll
            for (int i = 0; i < 8; ++i) { af[s][i] = lo[i]; af[s][i + 8] = hi[i]; }
        }

        // ---- 8 WMMAs: each B fragment reused across both A fragments ----
        #pragma unroll
        for (int j = 0; j < 4; ++j) {
            v16h bf = *(const v16h*)(&Bt[(j << 4) + nl][g << 4]);
            acc[0][j] = __builtin_amdgcn_wmma_f32_16x16x32_f16(
                false, af[0], false, bf, (short)0, acc[0][j], false, false);
            acc[1][j] = __builtin_amdgcn_wmma_f32_16x16x32_f16(
                false, af[1], false, bf, (short)0, acc[1][j], false, false);
        }
        __syncthreads();
    }

    // ---- epilogue: bias / relu / store (C layout: m = r + 8g, n = lane&15)
    #pragma unroll
    for (int j = 0; j < 4; ++j) {
        const int col = n0 + (j << 4) + nl;
        float bs = 0.f;
        if (HASBIAS) bs = bias[col];
        #pragma unroll
        for (int s = 0; s < 2; ++s) {
            #pragma unroll
            for (int r = 0; r < 8; ++r) {
                const int row = m0 + (wv << 5) + (s << 4) + (g << 3) + r;
                float v = acc[s][j][r] + bs;
                if (RELU) v = fmaxf(v, 0.f);
                float* op = Out + oOff + (size_t)row * ldOut + col;
                if (STREAM) __builtin_nontemporal_store(v, op);
                else        *op = v;
            }
        }
    }
}

// ---------------------------------------------------------------------------
// scale + optional causal mask + softmax over one score row
// ---------------------------------------------------------------------------
__global__ __launch_bounds__(256) void softmax_rows(
    float* __restrict__ S, int Lq, int Lk, float scale, int causal)
{
    const int l = blockIdx.x;
    const int z = blockIdx.y;
    float* row = S + ((size_t)z * Lq + l) * Lk;
    __shared__ float red[256];
    const int t = threadIdx.x;

    float mx = -1e30f;
    for (int m = t; m < Lk; m += 256) {
        float v = row[m] * scale;
        if (causal && m > l) v = -1e30f;
        mx = fmaxf(mx, v);
    }
    red[t] = mx; __syncthreads();
    for (int s = 128; s > 0; s >>= 1) {
        if (t < s) red[t] = fmaxf(red[t], red[t + s]);
        __syncthreads();
    }
    mx = red[0];
    __syncthreads();

    float sum = 0.f;
    for (int m = t; m < Lk; m += 256) {
        float e = 0.f;
        if (!(causal && m > l)) e = __expf(row[m] * scale - mx);
        row[m] = e;
        sum += e;
    }
    red[t] = sum; __syncthreads();
    for (int s = 128; s > 0; s >>= 1) {
        if (t < s) red[t] += red[t + s];
        __syncthreads();
    }
    const float inv = 1.f / red[0];
    for (int m = t; m < Lk; m += 256) row[m] *= inv;
}

// ---------------------------------------------------------------------------
// Out = LayerNorm(X + R) * gamma + beta   (row of D_MODEL; Out may alias X)
// ---------------------------------------------------------------------------
__global__ __launch_bounds__(256) void add_ln(
    const float* __restrict__ X, const float* __restrict__ R,
    const float* __restrict__ gamma, const float* __restrict__ beta,
    float* __restrict__ Out)
{
    const int row = blockIdx.x;
    const size_t off = (size_t)row * D_MODEL;
    const int t = threadIdx.x;
    __shared__ float red[256];

    float v[4];
    float s = 0.f;
    #pragma unroll
    for (int i = 0; i < 4; ++i) {
        const int c = t + (i << 8);
        v[i] = X[off + c] + R[off + c];
        s += v[i];
    }
    red[t] = s; __syncthreads();
    for (int st = 128; st > 0; st >>= 1) { if (t < st) red[t] += red[t + st]; __syncthreads(); }
    const float mu = red[0] * (1.f / D_MODEL);
    __syncthreads();

    float vs = 0.f;
    #pragma unroll
    for (int i = 0; i < 4; ++i) { const float d = v[i] - mu; vs += d * d; }
    red[t] = vs; __syncthreads();
    for (int st = 128; st > 0; st >>= 1) { if (t < st) red[t] += red[t + st]; __syncthreads(); }
    const float inv = rsqrtf(red[0] * (1.f / D_MODEL) + 1e-5f);

    #pragma unroll
    for (int i = 0; i < 4; ++i) {
        const int c = t + (i << 8);
        Out[off + c] = (v[i] - mu) * inv * gamma[c] + beta[c];
    }
}

// ---------------------------------------------------------------------------
// out[b, l, :] = emb[tok[l + tokOff, b], :] + positional_encoding(l, :)
// ---------------------------------------------------------------------------
__global__ __launch_bounds__(256) void embed_pe(
    const int* __restrict__ tok, const float* __restrict__ emb,
    float* __restrict__ out, int tokOff)
{
    const int l = blockIdx.x;
    const int b = blockIdx.y;
    const int t = threadIdx.x;
    const int id = tok[(size_t)(l + tokOff) * BATCH + b];
    const float* e = emb + (size_t)id * D_MODEL;
    float* o = out + ((size_t)b * SEQ + l) * D_MODEL;
    #pragma unroll
    for (int i = 0; i < 4; ++i) {
        const int j = t + (i << 8);
        const float ang = (float)l * __powf(10000.f, -(float)(j & ~1) * (1.f / D_MODEL));
        const float pe = (j & 1) ? __cosf(ang) : __sinf(ang);
        o[j] = e[j] + pe;
    }
}

// ---------------------------------------------------------------------------
// host-side helpers
// ---------------------------------------------------------------------------
template <int LAY, bool RELU, bool BIAS, bool STREAM = false>
static inline void rungemm(hipStream_t st, const float* A, const float* B,
                           const float* bias, float* O, int M, int N, int K,
                           int ldA, int ldB, int ldO, int batches = 1, int Hn = 1,
                           long long sAb = 0, long long sAh = 0,
                           long long sBb = 0, long long sBh = 0,
                           long long sOb = 0, long long sOh = 0)
{
    dim3 grid(N / 64, M / 128, batches);
    wmma_gemm<LAY, RELU, BIAS, STREAM><<<grid, dim3(128), 0, st>>>(
        A, B, bias, O, M, N, K, ldA, ldB, ldO, Hn, sAb, sAh, sBb, sBh, sOb, sOh);
}

extern "C" void kernel_launch(void* const* d_in, const int* in_sizes, int n_in,
                              void* d_out, int out_size, void* d_ws, size_t ws_size,
                              hipStream_t stream)
{
    (void)in_sizes; (void)n_in; (void)out_size; (void)ws_size;

    const int*   fr       = (const int*)d_in[0];
    const int*   en       = (const int*)d_in[1];
    const float* emb_fr   = (const float*)d_in[2];
    const float* emb_en   = (const float*)d_in[3];
    const float* e_qkv_w  = (const float*)d_in[4];
    const float* e_qkv_b  = (const float*)d_in[5];
    const float* e_f1w    = (const float*)d_in[6];
    const float* e_f1b    = (const float*)d_in[7];
    const float* e_f2w    = (const float*)d_in[8];
    const float* e_f2b    = (const float*)d_in[9];
    const float* e_lng    = (const float*)d_in[10];
    const float* e_lnb    = (const float*)d_in[11];
    const float* ds_qkv_w = (const float*)d_in[12];
    const float* ds_qkv_b = (const float*)d_in[13];
    const float* dc_qkv_w = (const float*)d_in[14];
    const float* dc_qkv_b = (const float*)d_in[15];
    const float* d_f1w    = (const float*)d_in[16];
    const float* d_f1b    = (const float*)d_in[17];
    const float* d_f2w    = (const float*)d_in[18];
    const float* d_f2b    = (const float*)d_in[19];
    const float* d_lng    = (const float*)d_in[20];
    const float* d_lnb    = (const float*)d_in[21];
    const float* out_w    = (const float*)d_in[22];
    const float* out_b    = (const float*)d_in[23];
    float* out = (float*)d_out;

    // workspace layout (floats)
    float* ws  = (float*)d_ws;
    const size_t MUL = (size_t)1 << 20;       // 1M floats
    float* xe  = ws;                          // encoder act / enc_out  (1M)
    float* q   = ws + 1 * MUL;                // (1M)
    float* kk  = ws + 2 * MUL;                // (1M)
    float* vv  = ws + 3 * MUL;                // (1M)
    float* att = ws + 4 * MUL;                // attn out / ffn h2      (1M)
    float* h1  = ws + 5 * MUL;                // ffn hidden             (1M)
    float* yd  = ws + 6 * MUL;                // decoder act            (1M)
    float* sc  = ws + 7 * MUL;                // scores B*H*L*L         (8M)

    const long long LD = (long long)SEQ * D_MODEL;  // 524288
    const long long LL = (long long)SEQ * SEQ;      // 262144
    const size_t PW = (size_t)NHEAD * D_MODEL * DHEAD;  // per-projection weight
    const size_t PB = (size_t)NHEAD * DHEAD;            // per-projection bias

    // embeddings + positional encoding
    embed_pe<<<dim3(SEQ, BATCH), 256, 0, stream>>>(fr, emb_fr, xe, 1);  // fr[1:]
    embed_pe<<<dim3(SEQ, BATCH), 256, 0, stream>>>(en, emb_en, yd, 0);

    auto mha = [&](const float* qin, const float* kvin, const float* w,
                   const float* b, int causal) {
        // QKV projections (head-blocked weight layout)
        rungemm<LAY_HDE, false, true>(stream, qin,  w + 0 * PW, b + 0 * PB, q,
                                      MROWS, D_MODEL, D_MODEL, D_MODEL, DHEAD, D_MODEL);
        rungemm<LAY_HDE, false, true>(stream, kvin, w + 1 * PW, b + 1 * PB, kk,
                                      MROWS, D_MODEL, D_MODEL, D_MODEL, DHEAD, D_MODEL);
        rungemm<LAY_HDE, false, true>(stream, kvin, w + 2 * PW, b + 2 * PB, vv,
                                      MROWS, D_MODEL, D_MODEL, D_MODEL, DHEAD, D_MODEL);
        // S = Q K^T  (batched over b,h)
        rungemm<LAY_NK, false, false>(stream, q, kk, nullptr, sc,
                                      SEQ, SEQ, DHEAD, D_MODEL, D_MODEL, SEQ,
                                      BATCH * NHEAD, NHEAD,
                                      LD, DHEAD, LD, DHEAD,
                                      (long long)NHEAD * LL, LL);
        softmax_rows<<<dim3(SEQ, BATCH * NHEAD), 256, 0, stream>>>(
            sc, SEQ, SEQ, 0.125f, causal);
        // O = P V
        rungemm<LAY_KN, false, false>(stream, sc, vv, nullptr, att,
                                      SEQ, DHEAD, SEQ, SEQ, D_MODEL, D_MODEL,
                                      BATCH * NHEAD, NHEAD,
                                      (long long)NHEAD * LL, LL,
                                      LD, DHEAD, LD, DHEAD);
    };

    auto ffn = [&](const float* act, const float* w1, const float* b1,
                   const float* w2, const float* b2) {
        rungemm<LAY_KN, true, true>(stream, act, w1, b1, h1,
                                    MROWS, D_MODEL, D_MODEL, D_MODEL, D_MODEL, D_MODEL);
        rungemm<LAY_KN, false, true>(stream, h1, w2, b2, att,
                                     MROWS, D_MODEL, D_MODEL, D_MODEL, D_MODEL, D_MODEL);
    };

    // ---------------- encoder ----------------
    for (int n = 0; n < NLAYERS; ++n) {
        mha(xe, xe, e_qkv_w + (size_t)n * 3 * PW, e_qkv_b + (size_t)n * 3 * PB, 0);
        add_ln<<<MROWS, 256, 0, stream>>>(xe, att,
            e_lng + (size_t)(n * 2 + 0) * D_MODEL,
            e_lnb + (size_t)(n * 2 + 0) * D_MODEL, xe);
        ffn(xe, e_f1w + (size_t)n * D_MODEL * D_MODEL, e_f1b + (size_t)n * D_MODEL,
                e_f2w + (size_t)n * D_MODEL * D_MODEL, e_f2b + (size_t)n * D_MODEL);
        add_ln<<<MROWS, 256, 0, stream>>>(xe, att,
            e_lng + (size_t)(n * 2 + 1) * D_MODEL,
            e_lnb + (size_t)(n * 2 + 1) * D_MODEL, xe);
    }
    // xe now holds enc_out

    // ---------------- decoder ----------------
    for (int n = 0; n < NLAYERS; ++n) {
        mha(yd, yd, ds_qkv_w + (size_t)n * 3 * PW, ds_qkv_b + (size_t)n * 3 * PB, 1);
        add_ln<<<MROWS, 256, 0, stream>>>(yd, att,
            d_lng + (size_t)(n * 3 + 0) * D_MODEL,
            d_lnb + (size_t)(n * 3 + 0) * D_MODEL, yd);
        mha(yd, xe, dc_qkv_w + (size_t)n * 3 * PW, dc_qkv_b + (size_t)n * 3 * PB, 0);
        add_ln<<<MROWS, 256, 0, stream>>>(yd, att,
            d_lng + (size_t)(n * 3 + 1) * D_MODEL,
            d_lnb + (size_t)(n * 3 + 1) * D_MODEL, yd);
        ffn(yd, d_f1w + (size_t)n * D_MODEL * D_MODEL, d_f1b + (size_t)n * D_MODEL,
                d_f2w + (size_t)n * D_MODEL * D_MODEL, d_f2b + (size_t)n * D_MODEL);
        add_ln<<<MROWS, 256, 0, stream>>>(yd, att,
            d_lng + (size_t)(n * 3 + 2) * D_MODEL,
            d_lnb + (size_t)(n * 3 + 2) * D_MODEL, yd);
    }

    // ---------------- vocab projection (streaming NT stores) ----------------
    rungemm<LAY_KN, false, true, true>(stream, yd, out_w, out_b, out,
                                       MROWS, VOCAB, D_MODEL, D_MODEL, VOCAB, VOCAB);
}
